// NewtonNet_75411035783765
// MI455X (gfx1250) — compile-verified, matrix operand order checked by reading, not run on hardware
//
#include <hip/hip_runtime.h>

#define FDIM   128
#define NNODES 16000
#define NEDGES 400000
#define NB     20
#define NLAYER 3
#define RCUT   5.0f

typedef __attribute__((ext_vector_type(16))) __bf16 v16bf;
typedef __attribute__((ext_vector_type(8)))  float  v8f;

#if __has_builtin(__builtin_amdgcn_tensor_load_to_lds)
typedef __attribute__((ext_vector_type(4))) unsigned tdm_g0_t;
typedef __attribute__((ext_vector_type(8))) int      tdm_g1_t;
typedef __attribute__((ext_vector_type(4))) int      tdm_g2_t;
typedef __attribute__((ext_vector_type(8))) int      tdm_g4_t;
#endif

// ---------- bf16 helpers (bit-level, round-to-nearest-even) ----------
__device__ __forceinline__ unsigned short f2bf(float x) {
  unsigned u = __float_as_uint(x);
  u += 0x7fffu + ((u >> 16) & 1u);
  return (unsigned short)(u >> 16);
}
__device__ __forceinline__ float bf2f(unsigned short h) {
  return __uint_as_float(((unsigned)h) << 16);
}
// Fast silu: result is rounded to bf16 downstream, so hardware v_rcp_f32
// (1 ulp) replaces the IEEE division sequence.
__device__ __forceinline__ float silu(float x) {
  return x * __builtin_amdgcn_rcpf(1.0f + __expf(-x));
}

// ---------- fragment loaders -----------------------------------------
// A fragment (16x32 bf16, ISA 7.12.2): lane i=l&15 is row M; this lane's 16
// elements are two contiguous 8-element runs at K = k0.. and K = k0+16..,
// k0 = kc + (l>=16 ? 8 : 0).  Two 16B vector loads + bit_cast.
struct P32 { uint4 a, b; };
__device__ __forceinline__ v16bf load_afrag(const unsigned short* A, int kc) {
  const int l = threadIdx.x & 31;
  const int i = l & 15;
  const int k0 = kc + ((l & 16) ? 8 : 0);
  P32 p;
  p.a = *(const uint4*)(A + i * FDIM + k0);
  p.b = *(const uint4*)(A + i * FDIM + k0 + 16);
  return __builtin_bit_cast(v16bf, p);
}

// B (weight) fragment from pre-swizzled fragment-major storage:
// wf[(((ntile*4 + kchunk)*32 + lane)<<4) + e] -> one 32B contiguous load.
__device__ __forceinline__ v16bf load_wfrag(const unsigned short* wf, int ntile, int kchunk) {
  const int lane = threadIdx.x & 31;
  const unsigned short* p = wf + ((((ntile << 2) + kchunk) * 32 + lane) << 4);
  return *(const v16bf*)p;
}

// Dual-M: two A tiles share each B fragment; the alternating independent
// accumulators also fill the WMMA->WMMA RAW hazard slots (ISA 7.12.1).
__device__ __forceinline__ void gemm128x2(const unsigned short* A0, const unsigned short* A1,
                                          const unsigned short* WF, int ntile,
                                          v8f& acc0, v8f& acc1) {
#pragma unroll
  for (int kc = 0; kc < 4; ++kc) {
    v16bf b  = load_wfrag(WF, ntile, kc);
    v16bf a0 = load_afrag(A0, kc * 32);
    v16bf a1 = load_afrag(A1, kc * 32);
    acc0 = __builtin_amdgcn_wmma_f32_16x16x32_bf16(false, a0, false, b, (short)0, acc0,
                                                   false, false);
    acc1 = __builtin_amdgcn_wmma_f32_16x16x32_bf16(false, a1, false, b, (short)0, acc1,
                                                   false, false);
  }
}

// ---------- weight swizzle: f32 [nmat][128][128] (K-major) -> bf16 frag-major
__global__ void k_swizzle_w(unsigned short* dst, const float* src, int nmat) {
  int idx = blockIdx.x * blockDim.x + threadIdx.x;
  if (idx >= nmat * FDIM * FDIM) return;
  int m = idx >> 14, r = idx & 16383;
  int fragid = r >> 9;        // (ntile,kchunk): 8*4
  int lane   = (r >> 4) & 31;
  int e      = r & 15;
  int ntile = fragid >> 2, kchunk = fragid & 3;
  int n = ntile * 16 + (lane & 15);
  int j = e >> 1, t = e & 1;
  int k = kchunk * 32 + ((lane & 16) ? 8 : 0) + ((j < 4) ? 2 * j : 16 + 2 * (j - 4)) + t;
  dst[idx] = f2bf(src[m * FDIM * FDIM + k * FDIM + n]);
}

// ---------- small utility kernels ------------------------------------
__global__ void k_zero(float* p, int n) {
  int i = blockIdx.x * blockDim.x + threadIdx.x;
  if (i < n) p[i] = 0.0f;
}

__global__ void k_init_atom(float* atom, const float* emb, const int* z) {
  int i = blockIdx.x * blockDim.x + threadIdx.x;
  if (i >= NNODES * FDIM) return;
  int n = i >> 7, f = i & (FDIM - 1);
  atom[i] = emb[z[n] * FDIM + f];
}

// dir_e [E,3], dist_e = fc * rbf [E,NB]
__global__ void k_edge_geom(const float* pos, const int* ei, const float* mu,
                            float* dir, float* de) {
  int e = blockIdx.x * blockDim.x + threadIdx.x;
  if (e >= NEDGES) return;
  int s = ei[e], d = ei[NEDGES + e];
  float dx = pos[s * 3 + 0] - pos[d * 3 + 0];
  float dy = pos[s * 3 + 1] - pos[d * 3 + 1];
  float dz = pos[s * 3 + 2] - pos[d * 3 + 2];
  float dist = fmaxf(sqrtf(dx * dx + dy * dy + dz * dz), 1e-8f);
  float inv = __builtin_amdgcn_rcpf(dist);
  dir[e * 3 + 0] = dx * inv;
  dir[e * 3 + 1] = dy * inv;
  dir[e * 3 + 2] = dz * inv;
  float fc = (dist < RCUT) ? 0.5f * (__cosf(3.14159265358979f * dist / RCUT) + 1.0f) : 0.0f;
  const float gamma = (NB / RCUT) * (NB / RCUT);  // 16
#pragma unroll
  for (int b = 0; b < NB; ++b) {
    float t = dist - mu[b];
    de[(size_t)e * NB + b] = fc * __expf(-gamma * t * t);
  }
}

// ---------- node MLP (mn), 32 nodes/block, dual-M -------------------
__global__ __launch_bounds__(256) void k_node_mn(const float* atom,
                                                 const unsigned short* W1, const float* b1,
                                                 const unsigned short* W2, const float* b2,
                                                 unsigned short* mn_bf) {
  __shared__ unsigned short sx[32 * FDIM];
  __shared__ unsigned short sh[32 * FDIM];
  const int node0 = blockIdx.x * 32;
  for (int i = threadIdx.x; i < 32 * FDIM; i += 256)
    sx[i] = f2bf(atom[(size_t)node0 * FDIM + i]);
  __syncthreads();

  const int wave = threadIdx.x >> 5, lane = threadIdx.x & 31;
  const int col = wave * 16 + (lane & 15), mrow = (lane & 16) ? 8 : 0;

  v8f a0, a1;
  float bias = b1[col];
#pragma unroll
  for (int r = 0; r < 8; ++r) { a0[r] = bias; a1[r] = bias; }
  gemm128x2(sx, sx + 16 * FDIM, W1, wave, a0, a1);
#pragma unroll
  for (int r = 0; r < 8; ++r) {
    sh[(mrow + r) * FDIM + col]      = f2bf(silu(a0[r]));
    sh[(16 + mrow + r) * FDIM + col] = f2bf(silu(a1[r]));
  }
  __syncthreads();

  v8f c0, c1;
  float bias2 = b2[col];
#pragma unroll
  for (int r = 0; r < 8; ++r) { c0[r] = bias2; c1[r] = bias2; }
  gemm128x2(sh, sh + 16 * FDIM, W2, wave, c0, c1);
#pragma unroll
  for (int r = 0; r < 8; ++r) {
    mn_bf[(size_t)(node0 + mrow + r) * FDIM + col]      = f2bf(c0[r]);
    mn_bf[(size_t)(node0 + 16 + mrow + r) * FDIM + col] = f2bf(c1[r]);
  }
}

// ---------- edge message: msg = (dist_e@meW) * mn[src] * mn[dst] ----
__global__ __launch_bounds__(256) void k_edge_msg(const float* de, const float* meW,
                                                  const unsigned short* mn_bf, const int* ei,
                                                  unsigned short* msg_bf, float* atom) {
  __shared__ float sde[2][NB];
  const int e = blockIdx.x * 2 + (threadIdx.x >> 7);
  const int f = threadIdx.x & (FDIM - 1);
  if (threadIdx.x < 2 * NB) {
    int el = threadIdx.x / NB, b = threadIdx.x % NB;
    sde[el][b] = de[(size_t)(blockIdx.x * 2 + el) * NB + b];
  }
  __syncthreads();
  const int s = ei[e], d = ei[NEDGES + e];
  float me = 0.0f;
#pragma unroll
  for (int b = 0; b < NB; ++b) me += sde[threadIdx.x >> 7][b] * meW[b * FDIM + f];
  float m = me * bf2f(mn_bf[(size_t)s * FDIM + f]) * bf2f(mn_bf[(size_t)d * FDIM + f]);
  msg_bf[(size_t)e * FDIM + f] = f2bf(m);
  atomicAdd(&atom[(size_t)s * FDIM + f], m);
}

// ---------- edge equivariant MLPs (dominant kernel, 32 edges/block) --
// o1 = silu(msg@W11)@W12 ; o2 = silu(msg@W21)@W22
// force_new[src] += o1 (x) dir_e + o2 * force_old[dst]
__global__ __launch_bounds__(256) void k_edge_eq(const unsigned short* msg_bf,
                                                 const unsigned short* W11,
                                                 const unsigned short* W12,
                                                 const unsigned short* W21,
                                                 const unsigned short* W22,
                                                 const float* dir, const int* ei,
                                                 const float* force_old, float* force_new) {
  __shared__ unsigned short sm[32 * FDIM];
  __shared__ unsigned short sh1[32 * FDIM];
  __shared__ unsigned short sh2[32 * FDIM];
  __shared__ float sdir[32][3];
  __shared__ int ss[32], sd[32];

  const int e0 = blockIdx.x * 32;

#if __has_builtin(__builtin_amdgcn_tensor_load_to_lds)
  // TDM: one wave issues a single 8KB DMA (1024 x 8B elements) of the msg
  // tile into LDS, tracked by TENSORcnt; other waves are held at the barrier.
  if (threadIdx.x < 32) {
    unsigned long long ga = (unsigned long long)(const void*)(msg_bf + (size_t)e0 * FDIM);
    unsigned lds = (unsigned)(unsigned long long)(void*)sm;  // generic addr[31:0] == LDS byte addr
    tdm_g0_t g0;
    g0[0] = 1u;                                    // count=1 user descriptor
    g0[1] = lds;                                   // lds_addr
    g0[2] = (unsigned)(ga & 0xFFFFFFFFu);          // global_addr[31:0]
    g0[3] = (unsigned)((ga >> 32) & 0x1FFFFFFu) | (2u << 30);  // addr[56:32] | type=2
    tdm_g1_t g1;
    g1[0] = (int)(3u << 16);                       // data_size = 8 bytes
    g1[1] = (int)(1024u << 16);                    // tensor_dim0 = 1024 (lo16 in [31:16])
    g1[2] = (int)(1u << 16);                       // tensor_dim0 hi = 0 ; tensor_dim1 = 1
    g1[3] = (int)(1024u << 16);                    // tensor_dim1 hi = 0 ; tile_dim0 = 1024
    g1[4] = 1;                                     // tile_dim1 = 1 ; tile_dim2 = 0
    g1[5] = 1024;                                  // tensor_dim0_stride lo32
    g1[6] = (int)(1024u << 16);                    // stride0 hi = 0 ; tensor_dim1_stride lo16
    g1[7] = 0;                                     // tensor_dim1_stride hi
    tdm_g2_t gz = {0, 0, 0, 0};
    tdm_g4_t gz8 = {0, 0, 0, 0, 0, 0, 0, 0};
    __builtin_amdgcn_tensor_load_to_lds(g0, g1, gz, gz, gz8, 0);
    __builtin_amdgcn_s_wait_tensorcnt(0);
  }
#else
  {  // fallback: manual bf16 tile copy as dwords
    const unsigned* g = (const unsigned*)(msg_bf + (size_t)e0 * FDIM);
    unsigned* l = (unsigned*)sm;
    for (int i = threadIdx.x; i < 32 * FDIM / 2; i += 256) l[i] = g[i];
  }
#endif
  if (threadIdx.x < 32) {
    ss[threadIdx.x] = ei[e0 + threadIdx.x];
    sd[threadIdx.x] = ei[NEDGES + e0 + threadIdx.x];
  }
  if (threadIdx.x < 96) ((float*)sdir)[threadIdx.x] = dir[(size_t)e0 * 3 + threadIdx.x];
  __syncthreads();

  const int wave = threadIdx.x >> 5, lane = threadIdx.x & 31;
  const int col = wave * 16 + (lane & 15), mrow = (lane & 16) ? 8 : 0;
  const unsigned short* A0 = sm;
  const unsigned short* A1 = sm + 16 * FDIM;

  v8f h1a = {}, h1b = {}, h2a = {}, h2b = {};
  gemm128x2(A0, A1, W11, wave, h1a, h1b);
  gemm128x2(A0, A1, W21, wave, h2a, h2b);
#pragma unroll
  for (int r = 0; r < 8; ++r) {
    sh1[(mrow + r) * FDIM + col]      = f2bf(silu(h1a[r]));
    sh1[(16 + mrow + r) * FDIM + col] = f2bf(silu(h1b[r]));
    sh2[(mrow + r) * FDIM + col]      = f2bf(silu(h2a[r]));
    sh2[(16 + mrow + r) * FDIM + col] = f2bf(silu(h2b[r]));
  }
  __syncthreads();

  v8f o1a = {}, o1b = {}, o2a = {}, o2b = {};
  gemm128x2(sh1, sh1 + 16 * FDIM, W12, wave, o1a, o1b);
  gemm128x2(sh2, sh2 + 16 * FDIM, W22, wave, o2a, o2b);

#pragma unroll
  for (int half = 0; half < 2; ++half) {
#pragma unroll
    for (int r = 0; r < 8; ++r) {
      const int m = half * 16 + mrow + r;
      const float v1 = half ? o1b[r] : o1a[r];
      const float v2 = half ? o2b[r] : o2a[r];
      const int s = ss[m], d = sd[m];
      const size_t sb = (size_t)s * 3 * FDIM + col;
      const size_t db = (size_t)d * 3 * FDIM + col;
#pragma unroll
      for (int dd = 0; dd < 3; ++dd) {
        float v = v1 * sdir[m][dd] + v2 * force_old[db + dd * FDIM];
        atomicAdd(&force_new[sb + dd * FDIM], v);
      }
    }
  }
}

// ---------- node update (iu), 32 nodes/block, dual-M ----------------
__global__ __launch_bounds__(256) void k_node_iu(float* atom,
                                                 const unsigned short* W1, const float* b1,
                                                 const unsigned short* W2, const float* b2,
                                                 const float* force) {
  __shared__ unsigned short sx[32 * FDIM];
  __shared__ unsigned short sh[32 * FDIM];
  const int node0 = blockIdx.x * 32;
  for (int i = threadIdx.x; i < 32 * FDIM; i += 256)
    sx[i] = f2bf(atom[(size_t)node0 * FDIM + i]);
  __syncthreads();

  const int wave = threadIdx.x >> 5, lane = threadIdx.x & 31;
  const int col = wave * 16 + (lane & 15), mrow = (lane & 16) ? 8 : 0;

  v8f a0, a1;
  float bias = b1[col];
#pragma unroll
  for (int r = 0; r < 8; ++r) { a0[r] = bias; a1[r] = bias; }
  gemm128x2(sx, sx + 16 * FDIM, W1, wave, a0, a1);
#pragma unroll
  for (int r = 0; r < 8; ++r) {
    sh[(mrow + r) * FDIM + col]      = f2bf(silu(a0[r]));
    sh[(16 + mrow + r) * FDIM + col] = f2bf(silu(a1[r]));
  }
  __syncthreads();

  v8f c0, c1;
  float bias2 = b2[col];
#pragma unroll
  for (int r = 0; r < 8; ++r) { c0[r] = bias2; c1[r] = bias2; }
  gemm128x2(sh, sh + 16 * FDIM, W2, wave, c0, c1);

#pragma unroll
  for (int half = 0; half < 2; ++half) {
#pragma unroll
    for (int r = 0; r < 8; ++r) {
      const int n = node0 + half * 16 + mrow + r;
      const float o = half ? c1[r] : c0[r];
      const size_t fb = (size_t)n * 3 * FDIM + col;
      float f0 = force[fb], f1 = force[fb + FDIM], f2v = force[fb + 2 * FDIM];
      float s2 = f0 * f0 + f1 * f1 + f2v * f2v;
      atom[(size_t)n * FDIM + col] += o * s2;
    }
  }
}

// ---------- host orchestration ---------------------------------------
extern "C" void kernel_launch(void* const* d_in, const int* in_sizes, int n_in,
                              void* d_out, int out_size, void* d_ws, size_t ws_size,
                              hipStream_t stream) {
  (void)in_sizes; (void)n_in; (void)out_size; (void)ws_size;
  const int*   z    = (const int*)d_in[0];
  const float* pos  = (const float*)d_in[1];
  const int*   ei   = (const int*)d_in[2];
  const float* emb  = (const float*)d_in[4];
  const float* mu   = (const float*)d_in[5];
  const float* mnW1 = (const float*)d_in[6];
  const float* mnb1 = (const float*)d_in[7];
  const float* mnW2 = (const float*)d_in[8];
  const float* mnb2 = (const float*)d_in[9];
  const float* meW  = (const float*)d_in[10];
  const float* e1W1 = (const float*)d_in[11];
  const float* e1W2 = (const float*)d_in[12];
  const float* e2W1 = (const float*)d_in[13];
  const float* e2W2 = (const float*)d_in[14];
  const float* iuW1 = (const float*)d_in[15];
  const float* iub1 = (const float*)d_in[16];
  const float* iuW2 = (const float*)d_in[17];
  const float* iub2 = (const float*)d_in[18];

  float* atom      = (float*)d_out;                       // [N,F]
  float* force_out = atom + (size_t)NNODES * FDIM;        // [N,3,F]

  char* ws = (char*)d_ws;
  size_t off = 0;
  auto carve = [&](size_t bytes) -> char* {
    size_t o = off;
    off = (off + bytes + 255) & ~(size_t)255;
    return ws + o;
  };

  const size_t WSZ = (size_t)NLAYER * FDIM * FDIM;  // elements per weight tensor
  unsigned short* w_mn1 = (unsigned short*)carve(WSZ * 2);
  unsigned short* w_mn2 = (unsigned short*)carve(WSZ * 2);
  unsigned short* w_e11 = (unsigned short*)carve(WSZ * 2);
  unsigned short* w_e12 = (unsigned short*)carve(WSZ * 2);
  unsigned short* w_e21 = (unsigned short*)carve(WSZ * 2);
  unsigned short* w_e22 = (unsigned short*)carve(WSZ * 2);
  unsigned short* w_iu1 = (unsigned short*)carve(WSZ * 2);
  unsigned short* w_iu2 = (unsigned short*)carve(WSZ * 2);
  unsigned short* mn_bf  = (unsigned short*)carve((size_t)NNODES * FDIM * 2);
  unsigned short* msg_bf = (unsigned short*)carve((size_t)NEDGES * FDIM * 2);
  float* dir = (float*)carve((size_t)NEDGES * 3 * 4);
  float* de  = (float*)carve((size_t)NEDGES * NB * 4);
  float* fA  = (float*)carve((size_t)NNODES * 3 * FDIM * 4);
  float* fB  = (float*)carve((size_t)NNODES * 3 * FDIM * 4);

  const int cn = (int)WSZ, cg = (cn + 255) / 256;
  k_swizzle_w<<<cg, 256, 0, stream>>>(w_mn1, mnW1, NLAYER);
  k_swizzle_w<<<cg, 256, 0, stream>>>(w_mn2, mnW2, NLAYER);
  k_swizzle_w<<<cg, 256, 0, stream>>>(w_e11, e1W1, NLAYER);
  k_swizzle_w<<<cg, 256, 0, stream>>>(w_e12, e1W2, NLAYER);
  k_swizzle_w<<<cg, 256, 0, stream>>>(w_e21, e2W1, NLAYER);
  k_swizzle_w<<<cg, 256, 0, stream>>>(w_e22, e2W2, NLAYER);
  k_swizzle_w<<<cg, 256, 0, stream>>>(w_iu1, iuW1, NLAYER);
  k_swizzle_w<<<cg, 256, 0, stream>>>(w_iu2, iuW2, NLAYER);

  k_init_atom<<<(NNODES * FDIM + 255) / 256, 256, 0, stream>>>(atom, emb, z);
  k_zero<<<(NNODES * 3 * FDIM + 255) / 256, 256, 0, stream>>>(fA, NNODES * 3 * FDIM);
  k_edge_geom<<<(NEDGES + 255) / 256, 256, 0, stream>>>(pos, ei, mu, dir, de);

  float* fcur = fA;
  float* fnxt = fB;
  for (int l = 0; l < NLAYER; ++l) {
    const size_t wo = (size_t)l * FDIM * FDIM;
    k_node_mn<<<NNODES / 32, 256, 0, stream>>>(atom, w_mn1 + wo, mnb1 + l * FDIM,
                                               w_mn2 + wo, mnb2 + l * FDIM, mn_bf);
    k_edge_msg<<<NEDGES / 2, 256, 0, stream>>>(de, meW + (size_t)l * NB * FDIM, mn_bf, ei,
                                               msg_bf, atom);
    (void)hipMemcpyAsync(fnxt, fcur, (size_t)NNODES * 3 * FDIM * 4,
                         hipMemcpyDeviceToDevice, stream);
    k_edge_eq<<<NEDGES / 32, 256, 0, stream>>>(msg_bf, w_e11 + wo, w_e12 + wo, w_e21 + wo,
                                               w_e22 + wo, dir, ei, fcur, fnxt);
    { float* t = fcur; fcur = fnxt; fnxt = t; }
    k_node_iu<<<NNODES / 32, 256, 0, stream>>>(atom, w_iu1 + wo, iub1 + l * FDIM,
                                               w_iu2 + wo, iub2 + l * FDIM, fcur);
  }
  (void)hipMemcpyAsync(force_out, fcur, (size_t)NNODES * 3 * FDIM * 4,
                       hipMemcpyDeviceToDevice, stream);
}
